// LinformerAttention_171798692543
// MI455X (gfx1250) — compile-verified, matrix operand order checked by reading, not run on hardware
//
#include <hip/hip_runtime.h>
#include <cstdint>
#include <cstddef>

// ---------- types ----------
typedef unsigned short u16;
typedef __attribute__((ext_vector_type(4)))  unsigned int uv4;   // 16-byte move
typedef __attribute__((ext_vector_type(16))) __bf16       v16bf; // WMMA A/B frag
typedef __attribute__((ext_vector_type(8)))  float        v8f;   // WMMA C/D frag

#define B_      4
#define SEQ_    4096
#define DIM_    1024
#define HEADS_  16
#define DH_     64
#define KLR_    256
#define MTOT_   (B_ * SEQ_)   // 16384

__device__ __forceinline__ u16 f2bf(float f) {
  unsigned u = __float_as_uint(f);
  u += 0x7FFFu + ((u >> 16) & 1u);   // round-to-nearest-even
  return (u16)(u >> 16);
}

// Async 16B global -> LDS copy (per lane), tracked by ASYNCcnt.
// Generic shared-pointer low 32 bits == wave-relative LDS byte offset.
__device__ __forceinline__ void async_ld16(const void* lds, const void* g) {
  unsigned lo = (unsigned)(uintptr_t)lds;
  unsigned long long ga = (unsigned long long)(uintptr_t)g;
  asm volatile("global_load_async_to_lds_b128 %0, %1, off"
               :: "v"(lo), "v"(ga)
               : "memory");
}
__device__ __forceinline__ void wait_async_le8() {
  asm volatile("s_wait_asynccnt 0x8" ::: "memory");
}
__device__ __forceinline__ void wait_async_0() {
  asm volatile("s_wait_asynccnt 0x0" ::: "memory");
}

// Load a 16x32 bf16 fragment from LDS following the CDNA5 16-bit A/B layout:
// lanes 0-15: row = row0+lane, K = k0 + {0..7, 16..23}
// lanes 16-31: row = row0+lane-16, K = k0 + {8..15, 24..31}
__device__ __forceinline__ v16bf lds_frag(const u16* base, int row0, int k0, int ld) {
  const int lane = threadIdx.x & 31;
  const u16* p = base + (row0 + (lane & 15)) * ld + k0 + ((lane & 16) ? 8 : 0);
  union { uv4 u[2]; v16bf v; } f;
  f.u[0] = *(const uv4*)(p);        // ds_load_b128
  f.u[1] = *(const uv4*)(p + 16);   // ds_load_b128
  return f.v;
}

__device__ __forceinline__ v8f wmma_bf16(v16bf a, v16bf b, v8f c) {
  return __builtin_amdgcn_wmma_f32_16x16x32_bf16(false, a, false, b, (short)0, c,
                                                 false, false);
}

// ---------- conversion kernels ----------
__global__ void cvt_bf16_kernel(const float* __restrict__ in, u16* __restrict__ out, int n) {
  int i = blockIdx.x * blockDim.x + threadIdx.x;
  if (i < n) out[i] = f2bf(in[i]);
}

// in: [R, C] f32 row-major -> out: [C, R] bf16 row-major (coalesced writes)
__global__ void cvt_t_bf16_kernel(const float* __restrict__ in, u16* __restrict__ out,
                                  int R, int C) {
  int i = blockIdx.x * blockDim.x + threadIdx.x;
  if (i < R * C) {
    int r = i % R, c = i / R;
    out[i] = f2bf(in[(size_t)r * C + c]);
  }
}

// ---------- generic bf16 WMMA GEMM (async double-buffered staging) ----------
// C[M,N] = A[M,K] * B'[N,K]^T ; A,B' bf16 row-major, K-contiguous.
// grid = (N/128, M/128), block = 256 (8 waves). Each wave: 32x64 sub-tile.
// mode 0: bf16 C[row*ldc+col]   mode 1: bf16 C[col*ldc+row] (vectorized)
// mode 2: f32 C[row*ldc+col] + bias
__global__ __launch_bounds__(256) void gemm_bf16_kernel(
    const u16* __restrict__ A, int lda,
    const u16* __restrict__ Bt, int ldb,
    int Kd, int mode, void* __restrict__ C, int ldc,
    const float* __restrict__ bias) {
  __shared__ __align__(16) u16 sA[2][128 * 72];   // 128x64, pad to 72
  __shared__ __align__(16) u16 sB[2][128 * 72];

  const int t    = threadIdx.x;
  const int lane = t & 31;
  const int w    = t >> 5;
  const int m0   = blockIdx.y * 128;
  const int n0   = blockIdx.x * 128;
  const int mo   = (w & 3) * 32;   // wave row offset in tile
  const int no   = (w >> 2) * 64;  // wave col offset in tile

  // 8 async b128 per thread per chunk (4 for A, 4 for B)
  auto issue_chunk = [&](int buf, int k0) {
#pragma unroll
    for (int i = 0; i < 4; ++i) {
      int c   = t + i * 256;
      int row = c >> 3;
      int seg = (c & 7) * 8;
      async_ld16(&sA[buf][row * 72 + seg], &A[(size_t)(m0 + row) * lda + k0 + seg]);
      async_ld16(&sB[buf][row * 72 + seg], &Bt[(size_t)(n0 + row) * ldb + k0 + seg]);
    }
  };

  v8f zero = {0.f, 0.f, 0.f, 0.f, 0.f, 0.f, 0.f, 0.f};
  v8f acc[2][4];
#pragma unroll
  for (int i = 0; i < 2; ++i)
#pragma unroll
    for (int j = 0; j < 4; ++j) acc[i][j] = zero;

  const int nch = Kd >> 6;   // 64-wide K chunks
  issue_chunk(0, 0);
  for (int ci = 0; ci < nch; ++ci) {
    if (ci + 1 < nch) {
      issue_chunk((ci + 1) & 1, (ci + 1) * 64);  // prefetch next chunk
      wait_async_le8();                          // this wave's chunk ci landed
    } else {
      wait_async_0();
    }
    __syncthreads();                             // everyone's chunk ci landed
    const u16* cA = sA[ci & 1];
    const u16* cB = sB[ci & 1];
#pragma unroll
    for (int ks = 0; ks < 2; ++ks) {
      v16bf a0 = lds_frag(cA, mo, ks * 32, 72);
      v16bf a1 = lds_frag(cA, mo + 16, ks * 32, 72);
#pragma unroll
      for (int j = 0; j < 4; ++j) {
        v16bf b = lds_frag(cB, no + j * 16, ks * 32, 72);
        acc[0][j] = wmma_bf16(a0, b, acc[0][j]);
        acc[1][j] = wmma_bf16(a1, b, acc[1][j]);
      }
    }
    __syncthreads();                             // buffer safe to overwrite
  }

  // epilogue: C frag layout -> lane 0-15: M=i, N=lane ; lane 16-31: M=8+i
  const int hi = (lane >> 4) * 8;
  const int cl = lane & 15;
#pragma unroll
  for (int i2 = 0; i2 < 2; ++i2)
#pragma unroll
    for (int j = 0; j < 4; ++j) {
      int rb = m0 + mo + i2 * 16 + hi;   // multiple of 8
      int cb = n0 + no + j * 16 + cl;
      if (mode == 1) {
        // transposed: 8 consecutive rows per lane -> one b128 store
        union { uv4 u; u16 h[8]; } pk;
#pragma unroll
        for (int i = 0; i < 8; ++i) pk.h[i] = f2bf(acc[i2][j][i]);
        *(uv4*)&((u16*)C)[(size_t)cb * ldc + rb] = pk.u;
      } else if (mode == 0) {
#pragma unroll
        for (int i = 0; i < 8; ++i)
          ((u16*)C)[(size_t)(rb + i) * ldc + cb] = f2bf(acc[i2][j][i]);
      } else {
#pragma unroll
        for (int i = 0; i < 8; ++i)
          ((float*)C)[(size_t)(rb + i) * ldc + cb] = acc[i2][j][i] + bias[cb];
      }
    }
}

// ---------- fused Linformer attention ----------
// q: bf16 [B*N, 1024]; kp: bf16 [B][256,1024]; vpT: bf16 [B][1024,256]
// ao: bf16 [B*N, 1024]. grid = (N/64, B*H), block = 256.
__global__ __launch_bounds__(256) void linformer_attn_kernel(
    const u16* __restrict__ qb, const u16* __restrict__ kp,
    const u16* __restrict__ vpT, u16* __restrict__ ao) {
  // LDS: [0,9216) q | [9216,46080) k -- aliased by dots f32 [0,65536)
  //      [65536,99328) attn bf16 ; v bf16 [0,33792) aliases dots after softmax
  __shared__ __align__(16) char smem[99328];
  u16*   s_q = (u16*)smem;                // 64 x 72
  u16*   s_k = (u16*)(smem + 9216);       // 256 x 72
  float* s_d = (float*)smem;              // 64 x 256 f32
  u16*   s_a = (u16*)(smem + 65536);      // 64 x 264 bf16
  u16*   s_v = (u16*)smem;                // 64 x 264 bf16

  const int t    = threadIdx.x;
  const int lane = t & 31;
  const int w    = t >> 5;
  const int n0   = blockIdx.x * 64;
  const int bh   = blockIdx.y;
  const int b    = bh >> 4;
  const int h    = bh & 15;

  const u16* qg = qb + (size_t)(b * SEQ_ + n0) * DIM_ + h * DH_;
  const u16* kg = kp + (size_t)b * KLR_ * DIM_ + h * DH_;

  // stage q tile [64,64] and k head [256,64] via async copies
#pragma unroll
  for (int i = 0; i < 2; ++i) {
    int c = t + i * 256, row = c >> 3, seg = (c & 7) * 8;
    async_ld16(&s_q[row * 72 + seg], &qg[(size_t)row * DIM_ + seg]);
  }
#pragma unroll
  for (int i = 0; i < 8; ++i) {
    int c = t + i * 256, row = c >> 3, seg = (c & 7) * 8;
    async_ld16(&s_k[row * 72 + seg], &kg[(size_t)row * DIM_ + seg]);
  }
  wait_async_0();
  __syncthreads();

  // dots = q @ k^T * 0.125 : 64x256, each wave 16 rows x 128 cols (8 frags)
  const int rw = (w >> 1) * 16;
  const int cw = (w & 1) * 128;
  v8f zero = {0.f, 0.f, 0.f, 0.f, 0.f, 0.f, 0.f, 0.f};
  v8f acc[8];
#pragma unroll
  for (int j = 0; j < 8; ++j) acc[j] = zero;
#pragma unroll
  for (int ks = 0; ks < 2; ++ks) {
    v16bf a = lds_frag(s_q, rw, ks * 32, 72);
#pragma unroll
    for (int j = 0; j < 8; ++j) {
      v16bf bf = lds_frag(s_k, cw + j * 16, ks * 32, 72);
      acc[j] = wmma_bf16(a, bf, acc[j]);
    }
  }
  __syncthreads();  // q/k dead; safe to overwrite with dots

  const int hi = (lane >> 4) * 8;
  const int cl = lane & 15;
#pragma unroll
  for (int j = 0; j < 8; ++j)
#pragma unroll
    for (int i = 0; i < 8; ++i)
      s_d[(rw + i + hi) * 256 + cw + j * 16 + cl] = acc[j][i] * 0.125f;
  __syncthreads();

  // softmax over 256 per row (one thread per row)
  if (t < 64) {
    float mx = -1e30f;
    for (int c = 0; c < 256; ++c) mx = fmaxf(mx, s_d[t * 256 + c]);
    float sum = 0.f;
    for (int c = 0; c < 256; ++c) {
      float e = __expf(s_d[t * 256 + c] - mx);
      s_d[t * 256 + c] = e;
      sum += e;
    }
    float r = 1.f / sum;
    for (int c = 0; c < 256; ++c) s_a[t * 264 + c] = f2bf(s_d[t * 256 + c] * r);
  }
  __syncthreads();  // dots dead; safe to overwrite with v

  // stage v head async: vpT rows [h*64, h*64+64) x 256
  const u16* vg = vpT + (size_t)b * DIM_ * KLR_ + (size_t)h * DH_ * KLR_;
#pragma unroll
  for (int i = 0; i < 8; ++i) {
    int c = t + i * 256, row = c >> 5, seg = (c & 31) * 8;
    async_ld16(&s_v[row * 264 + seg], &vg[(size_t)row * KLR_ + seg]);
  }
  wait_async_0();
  __syncthreads();

  // out = attn @ v : 64x64, each wave 16 rows x 32 cols (2 frags), K=256
  const int co = (w & 1) * 32;
  v8f o[2];
  o[0] = zero; o[1] = zero;
#pragma unroll
  for (int kk = 0; kk < 8; ++kk) {
    v16bf a = lds_frag(s_a, rw, kk * 32, 264);
#pragma unroll
    for (int j = 0; j < 2; ++j) {
      v16bf bf = lds_frag(s_v, co + j * 16, kk * 32, 264);
      o[j] = wmma_bf16(a, bf, o[j]);
    }
  }
  u16* og = ao + (size_t)(b * SEQ_ + n0) * DIM_ + h * DH_;
#pragma unroll
  for (int j = 0; j < 2; ++j)
#pragma unroll
    for (int i = 0; i < 8; ++i)
      og[(size_t)(rw + i + hi) * DIM_ + co + j * 16 + cl] = f2bf(o[j][i]);
}

// ---------- host ----------
extern "C" void kernel_launch(void* const* d_in, const int* in_sizes, int n_in,
                              void* d_out, int out_size, void* d_ws, size_t ws_size,
                              hipStream_t stream) {
  const float* x   = (const float*)d_in[0];
  const float* Wq  = (const float*)d_in[1];
  const float* Wk  = (const float*)d_in[2];
  const float* Wv  = (const float*)d_in[3];
  const float* pk  = (const float*)d_in[4];
  const float* pv  = (const float*)d_in[5];
  const float* Wo  = (const float*)d_in[6];
  const float* bo  = (const float*)d_in[7];
  float* out = (float*)d_out;

  char* ws = (char*)d_ws;
  size_t off = 0;
  auto alloc = [&](size_t bytes) -> void* {
    void* p = ws + off;
    off += (bytes + 255) & ~(size_t)255;
    return p;
  };

  const size_t NX = (size_t)MTOT_ * DIM_;          // 16M elems
  u16* xb  = (u16*)alloc(NX * 2);                  // x bf16
  u16* wqb = (u16*)alloc((size_t)DIM_ * DIM_ * 2);
  u16* wkb = (u16*)alloc((size_t)DIM_ * DIM_ * 2);
  u16* wvb = (u16*)alloc((size_t)DIM_ * DIM_ * 2);
  u16* wob = (u16*)alloc((size_t)DIM_ * DIM_ * 2);
  u16* pkT = (u16*)alloc((size_t)KLR_ * SEQ_ * 2); // [256,4096]
  u16* pvT = (u16*)alloc((size_t)KLR_ * SEQ_ * 2);
  u16* qb  = (u16*)alloc(NX * 2);                  // q bf16 [16384,1024]
  u16* kT  = (u16*)alloc(NX * 2);                  // keys^T bf16 [1024,16384]
  u16* vT  = (u16*)alloc(NX * 2);                  // vals^T bf16 [1024,16384]
  u16* kp  = (u16*)alloc((size_t)B_ * KLR_ * DIM_ * 2); // keys_p [b][256,1024]
  u16* vpT = (u16*)alloc((size_t)B_ * DIM_ * KLR_ * 2); // vals_p^T [b][1024,256]
  u16* ao  = xb;  // x bf16 dead after QKV GEMMs; reuse for attention output

  // 1) conversions
  cvt_bf16_kernel<<<(int)((NX + 255) / 256), 256, 0, stream>>>(x, xb, (int)NX);
  cvt_bf16_kernel<<<(DIM_ * DIM_ + 255) / 256, 256, 0, stream>>>(Wq, wqb, DIM_ * DIM_);
  cvt_bf16_kernel<<<(DIM_ * DIM_ + 255) / 256, 256, 0, stream>>>(Wk, wkb, DIM_ * DIM_);
  cvt_bf16_kernel<<<(DIM_ * DIM_ + 255) / 256, 256, 0, stream>>>(Wv, wvb, DIM_ * DIM_);
  cvt_bf16_kernel<<<(DIM_ * DIM_ + 255) / 256, 256, 0, stream>>>(Wo, wob, DIM_ * DIM_);
  cvt_t_bf16_kernel<<<(SEQ_ * KLR_ + 255) / 256, 256, 0, stream>>>(pk, pkT, SEQ_, KLR_);
  cvt_t_bf16_kernel<<<(SEQ_ * KLR_ + 255) / 256, 256, 0, stream>>>(pv, pvT, SEQ_, KLR_);

  // 2) QKV projections (B' = W since weights are [out,in])
  dim3 g1(DIM_ / 128, MTOT_ / 128);  // (8, 128)
  gemm_bf16_kernel<<<g1, 256, 0, stream>>>(xb, DIM_, wqb, DIM_, DIM_, 0, qb, DIM_, nullptr);
  gemm_bf16_kernel<<<g1, 256, 0, stream>>>(xb, DIM_, wkb, DIM_, DIM_, 1, kT, MTOT_, nullptr);
  gemm_bf16_kernel<<<g1, 256, 0, stream>>>(xb, DIM_, wvb, DIM_, DIM_, 1, vT, MTOT_, nullptr);

  // 3) low-rank sequence projections per batch: [256,4096] x [4096,1024]
  dim3 g2(DIM_ / 128, KLR_ / 128);   // (8, 2)
  for (int b = 0; b < B_; ++b) {
    gemm_bf16_kernel<<<g2, 256, 0, stream>>>(pkT, SEQ_, kT + (size_t)b * SEQ_, MTOT_,
                                             SEQ_, 0, kp + (size_t)b * KLR_ * DIM_, DIM_,
                                             nullptr);
    gemm_bf16_kernel<<<g2, 256, 0, stream>>>(pvT, SEQ_, vT + (size_t)b * SEQ_, MTOT_,
                                             SEQ_, 1, vpT + (size_t)b * DIM_ * KLR_, KLR_,
                                             nullptr);
  }

  // 4) fused attention (dots -> softmax -> out), per (b,h,64-row tile)
  dim3 ga(SEQ_ / 64, B_ * HEADS_);   // (64, 64)
  linformer_attn_kernel<<<ga, 256, 0, stream>>>(qb, kp, vpT, ao);

  // 5) output projection with bias, f32 out
  gemm_bf16_kernel<<<g1, 256, 0, stream>>>(ao, DIM_, wob, DIM_, DIM_, 2, out, DIM_, bo);

  (void)in_sizes; (void)n_in; (void)out_size; (void)ws_size;
}
